// LocalAttention_61778809585664
// MI455X (gfx1250) — matrix operand order repaired and checked by reading, not compiled, not run on hardware
//
#include <hip/hip_runtime.h>
#include <hip/hip_bf16.h>

#define D_MODEL   256
#define N_HEADS   8
#define HEAD_DIM  32
#define KNB       16
#define NPTS      4096
#define SCALE_F   0.17677669529663687f   // 1/sqrt(32)

typedef __bf16 bf16_t;
typedef bf16_t v16bf __attribute__((ext_vector_type(16)));
typedef bf16_t v8bf  __attribute__((ext_vector_type(8)));
typedef float  v8f   __attribute__((ext_vector_type(8)));

static __device__ __forceinline__ bf16_t f2bf(float f) {
    // round-to-nearest-even fp32 -> bf16
    unsigned int u = __builtin_bit_cast(unsigned int, f);
    u += 0x7FFFu + ((u >> 16) & 1u);
    unsigned short h = (unsigned short)(u >> 16);
    return __builtin_bit_cast(bf16_t, h);
}

// ---------------------------------------------------------------------------
// Bulk fp32 -> bf16 conversion (8 elements / thread, 16B stores)
// ---------------------------------------------------------------------------
__global__ __launch_bounds__(256)
void cvt_f32_to_bf16(const float* __restrict__ in, bf16_t* __restrict__ out, int n8)
{
    const int t = blockIdx.x * 256 + threadIdx.x;
    if (t >= n8) return;
    const float4* p = reinterpret_cast<const float4*>(in) + (size_t)t * 2;
    const float4 a = p[0], b = p[1];
    v8bf o;
    o[0] = f2bf(a.x); o[1] = f2bf(a.y); o[2] = f2bf(a.z); o[3] = f2bf(a.w);
    o[4] = f2bf(b.x); o[5] = f2bf(b.y); o[6] = f2bf(b.z); o[7] = f2bf(b.w);
    reinterpret_cast<v8bf*>(out)[t] = o;
}

// ---------------------------------------------------------------------------
// Weight prep: W[N][K] fp32 (row-major, y = x @ W^T)  ->  Wt[K][N] bf16.
// After this, a WMMA B fragment element (lane = N column, fixed K) is one
// contiguous 32B read across 16 lanes per 16-column tile.
// ---------------------------------------------------------------------------
__global__ __launch_bounds__(256)
void w_transpose_bf16(const float* __restrict__ W, bf16_t* __restrict__ Wt)
{
    __shared__ float t[16][17];
    const int tx = threadIdx.x & 15, ty = threadIdx.x >> 4;
    const int n0 = blockIdx.x * 16, k0 = blockIdx.y * 16;
    t[ty][tx] = W[(size_t)(n0 + ty) * D_MODEL + k0 + tx];
    __syncthreads();
    Wt[(size_t)(k0 + ty) * D_MODEL + n0 + tx] = f2bf(t[tx][ty]);
}

// ---------------------------------------------------------------------------
// GEMM:  Y[M,256] = Xb[M,256](bf16) @ Wt[256,256](bf16, pre-swizzled) + bias
// One wave -> 16x64 output tile (4 accumulators share one A fragment).
// K loop fully unrolled: 32 v_wmma_f32_16x16x32_bf16 per wave, loads only.
// ---------------------------------------------------------------------------
__global__ __launch_bounds__(256)
void gemm_bf16_wmma(const bf16_t* __restrict__ Xb, const bf16_t* __restrict__ Wt,
                    const float* __restrict__ bias, float* __restrict__ Y, int M)
{
    constexpr int K = D_MODEL, N = D_MODEL;
    const int lane = threadIdx.x & 31;
    const int wave = threadIdx.x >> 5;
    const int nColTiles = N / 64;                 // 4 tiles of 64 columns
    const int tile = blockIdx.x * 8 + wave;
    const int mt = tile / nColTiles;
    const int ct = tile - mt * nColTiles;
    if (mt * 16 >= M) return;                     // wave-uniform: EXEC all-ones

    const int half = lane >> 4;                   // 0 | 1
    const int l15  = lane & 15;
    const int m0 = mt * 16, n0 = ct * 64;

    const bf16_t* arow = Xb + (size_t)(m0 + l15) * K;   // A: lane -> row M
    const int     kbl  = l15 + (half << 4);             // unused (B lane holds a column N)

    v8f acc[4] = {{}, {}, {}, {}};
    #pragma unroll
    for (int k0 = 0; k0 < K; k0 += 32) {
        // A fragment (16x32): lanes 0-15 hold K {0..7,16..23}, lanes 16-31
        // hold K {8..15,24..31} -> two 16B loads, register concat.
        const v8bf alo = *reinterpret_cast<const v8bf*>(arow + k0 + half * 8);
        const v8bf ahi = *reinterpret_cast<const v8bf*>(arow + k0 + 16 + half * 8);
        const v16bf a = __builtin_shufflevector(alo, ahi,
            0, 1, 2, 3, 4, 5, 6, 7, 8, 9, 10, 11, 12, 13, 14, 15);

        // B fragments: lane holds column N = n0+c*16+l15, K in the same order as A.
        const bf16_t* wcol = Wt + (size_t)k0 * N + n0 + l15;
        #pragma unroll
        for (int c = 0; c < 4; ++c) {
            v16bf b;
            #pragma unroll
            for (int s = 0; s < 16; ++s)
                b[s] = wcol[(size_t)((s & 7) + half * 8 + (s >> 3) * 16) * N + c * 16];
            acc[c] = __builtin_amdgcn_wmma_f32_16x16x32_bf16(
                         false, a, false, b, (short)0, acc[c], false, false);
        }
    }

    // C/D layout: VGPR j, lanes 0-15 -> M=j, lanes 16-31 -> M=8+j
    #pragma unroll
    for (int c = 0; c < 4; ++c) {
        const int col = n0 + c * 16 + l15;
        const float bb = bias[col];
        float* yp = Y + (size_t)(m0 + 8 * half) * N + col;
        #pragma unroll
        for (int j = 0; j < 8; ++j)
            yp[(size_t)j * N] = acc[c][j] + bb;
    }
}

// ---------------------------------------------------------------------------
// KNN: one thread per query point; candidate coords staged through LDS in
// 256-point tiles; register-resident sorted top-16 (static-index shifts).
// ---------------------------------------------------------------------------
__global__ __launch_bounds__(256)
void knn_kernel(const float* __restrict__ coord, int* __restrict__ idx_out, int N)
{
    __shared__ float sx[256], sy[256], sz[256];
    const int b = blockIdx.y;
    const int p = blockIdx.x * 256 + threadIdx.x;
    const float* cb = coord + (size_t)b * N * 3;

    const float qx = cb[p * 3 + 0];
    const float qy = cb[p * 3 + 1];
    const float qz = cb[p * 3 + 2];

    float bd[KNB]; int bi[KNB];
    #pragma unroll
    for (int i = 0; i < KNB; ++i) { bd[i] = 3.4e38f; bi[i] = 0; }

    for (int t0 = 0; t0 < N; t0 += 256) {
        __syncthreads();
        const int j = t0 + threadIdx.x;           // N % 256 == 0
        sx[threadIdx.x] = cb[j * 3 + 0];
        sy[threadIdx.x] = cb[j * 3 + 1];
        sz[threadIdx.x] = cb[j * 3 + 2];
        __syncthreads();
        for (int jj = 0; jj < 256; ++jj) {
            const float dx = qx - sx[jj];
            const float dy = qy - sy[jj];
            const float dz = qz - sz[jj];
            float cd = dx * dx + dy * dy + dz * dz;
            if (cd < bd[KNB - 1]) {
                int ci = t0 + jj;
                #pragma unroll
                for (int s = 0; s < KNB; ++s) {   // bubble insert, keeps sorted
                    if (cd < bd[s]) {
                        float td = bd[s]; bd[s] = cd; cd = td;
                        int   ti = bi[s]; bi[s] = ci; ci = ti;
                    }
                }
            }
        }
    }

    int* op = idx_out + ((size_t)b * N + p) * KNB;
    #pragma unroll
    for (int i = 0; i < KNB; ++i) op[i] = bi[i];
}

// ---------------------------------------------------------------------------
// Local attention: one thread per (point, head). 16-neighbor scores, softmax,
// weighted V sum. Gathers are L2-resident; V rows prefetched during scoring.
// Output written directly in bf16 (A operand of the O-projection GEMM).
// ---------------------------------------------------------------------------
__global__ __launch_bounds__(256)
void local_attn_kernel(const float* __restrict__ q, const float* __restrict__ kf,
                       const float* __restrict__ v, const int* __restrict__ idx,
                       bf16_t* __restrict__ outb, int BN, int N)
{
    const int t = blockIdx.x * blockDim.x + threadIdx.x;
    if (t >= BN * N_HEADS) return;
    const int head = t & (N_HEADS - 1);
    const int pt   = t >> 3;                  // b*N + n
    const int b    = pt / N;

    const size_t qoff = (size_t)pt * D_MODEL + head * HEAD_DIM;
    const float4* qp = reinterpret_cast<const float4*>(q + qoff);
    float4 qv[8];
    #pragma unroll
    for (int i = 0; i < 8; ++i) qv[i] = qp[i];

    const int* nbr = idx + (size_t)pt * KNB;
    int nbi[KNB];
    #pragma unroll
    for (int j = 0; j < KNB; ++j) nbi[j] = nbr[j];

    float sc[KNB];
    float mx = -3.4e38f;
    #pragma unroll
    for (int j = 0; j < KNB; ++j) {
        const size_t roff = ((size_t)(b * N + nbi[j])) * D_MODEL + head * HEAD_DIM;
        const float4* kp = reinterpret_cast<const float4*>(kf + roff);
        __builtin_prefetch(v + roff, 0, 1);   // global_prefetch for the V pass
        float sX = 0.f, sY = 0.f, sZ = 0.f, sW = 0.f;
        #pragma unroll
        for (int i = 0; i < 8; ++i) {
            const float4 kk = kp[i];
            sX += qv[i].x * kk.x; sY += qv[i].y * kk.y;
            sZ += qv[i].z * kk.z; sW += qv[i].w * kk.w;
        }
        const float s = (sX + sY + sZ + sW) * SCALE_F;
        sc[j] = s;
        mx = fmaxf(mx, s);
    }

    float sum = 0.f;
    #pragma unroll
    for (int j = 0; j < KNB; ++j) { sc[j] = __expf(sc[j] - mx); sum += sc[j]; }
    const float inv = 1.0f / sum;

    float4 acc[8] = {};
    #pragma unroll
    for (int j = 0; j < KNB; ++j) {
        const size_t roff = ((size_t)(b * N + nbi[j])) * D_MODEL + head * HEAD_DIM;
        const float4* vp = reinterpret_cast<const float4*>(v + roff);
        const float w = sc[j] * inv;
        #pragma unroll
        for (int i = 0; i < 8; ++i) {
            const float4 vv = vp[i];
            acc[i].x += w * vv.x; acc[i].y += w * vv.y;
            acc[i].z += w * vv.z; acc[i].w += w * vv.w;
        }
    }

    v8bf* op = reinterpret_cast<v8bf*>(outb + qoff);
    #pragma unroll
    for (int c = 0; c < 4; ++c) {
        const float4 lo = acc[2 * c], hi = acc[2 * c + 1];
        v8bf o;
        o[0] = f2bf(lo.x); o[1] = f2bf(lo.y); o[2] = f2bf(lo.z); o[3] = f2bf(lo.w);
        o[4] = f2bf(hi.x); o[5] = f2bf(hi.y); o[6] = f2bf(hi.z); o[7] = f2bf(hi.w);
        op[c] = o;
    }
}

// ---------------------------------------------------------------------------
extern "C" void kernel_launch(void* const* d_in, const int* in_sizes, int n_in,
                              void* d_out, int out_size, void* d_ws, size_t ws_size,
                              hipStream_t stream)
{
    const float* x     = (const float*)d_in[0];
    const float* coord = (const float*)d_in[1];
    const float* Wq    = (const float*)d_in[2];
    const float* bq    = (const float*)d_in[3];
    const float* Wk    = (const float*)d_in[4];
    const float* bk    = (const float*)d_in[5];
    const float* Wv    = (const float*)d_in[6];
    const float* bvv   = (const float*)d_in[7];
    const float* Wo    = (const float*)d_in[8];
    const float* bo    = (const float*)d_in[9];
    float* out = (float*)d_out;

    const int BN = in_sizes[0] / D_MODEL;     // B * N
    const int N  = NPTS;
    const int B  = BN / N;

    // ---- workspace layout (byte offsets, all generously aligned) ----
    char* ws = (char*)d_ws;
    size_t off = 0;
    int* nidx = (int*)(ws + off);           off += (size_t)BN * KNB * sizeof(int);
    bf16_t* xbf = (bf16_t*)(ws + off);      off += (size_t)BN * D_MODEL * sizeof(bf16_t);
    bf16_t* wtq = (bf16_t*)(ws + off);      off += (size_t)D_MODEL * D_MODEL * sizeof(bf16_t);
    bf16_t* wtk = (bf16_t*)(ws + off);      off += (size_t)D_MODEL * D_MODEL * sizeof(bf16_t);
    bf16_t* wtv = (bf16_t*)(ws + off);      off += (size_t)D_MODEL * D_MODEL * sizeof(bf16_t);
    bf16_t* wto = (bf16_t*)(ws + off);      off += (size_t)D_MODEL * D_MODEL * sizeof(bf16_t);
    float* qbuf = (float*)(ws + off);       off += (size_t)BN * D_MODEL * sizeof(float);
    float* kbuf = (float*)(ws + off);       off += (size_t)BN * D_MODEL * sizeof(float);
    float* vbuf = (float*)(ws + off);       off += (size_t)BN * D_MODEL * sizeof(float);
    bf16_t* abf = (bf16_t*)(ws + off);      off += (size_t)BN * D_MODEL * sizeof(bf16_t);

    // ---- operand preparation (bf16 activations, pre-swizzled bf16 weights) ----
    const int n8 = BN * D_MODEL / 8;
    cvt_f32_to_bf16<<<(n8 + 255) / 256, 256, 0, stream>>>(x, xbf, n8);
    dim3 wgrid(D_MODEL / 16, D_MODEL / 16);
    w_transpose_bf16<<<wgrid, 256, 0, stream>>>(Wq, wtq);
    w_transpose_bf16<<<wgrid, 256, 0, stream>>>(Wk, wtk);
    w_transpose_bf16<<<wgrid, 256, 0, stream>>>(Wv, wtv);
    w_transpose_bf16<<<wgrid, 256, 0, stream>>>(Wo, wto);

    // ---- KNN (independent of projections) ----
    dim3 kgrid(N / 256, B);
    knn_kernel<<<kgrid, 256, 0, stream>>>(coord, nidx, N);

    // ---- Q/K/V projections: 16x64 tile per wave, 8 waves per block ----
    const int nTiles  = (BN / 16) * (D_MODEL / 64);   // 4096
    const int gblocks = nTiles / 8;                   // 512
    gemm_bf16_wmma<<<gblocks, 256, 0, stream>>>(xbf, wtq, bq,  qbuf, BN);
    gemm_bf16_wmma<<<gblocks, 256, 0, stream>>>(xbf, wtk, bk,  kbuf, BN);
    gemm_bf16_wmma<<<gblocks, 256, 0, stream>>>(xbf, wtv, bvv, vbuf, BN);

    // ---- local attention (writes bf16 A operand for O-projection) ----
    const int athreads = BN * N_HEADS;
    local_attn_kernel<<<(athreads + 255) / 256, 256, 0, stream>>>(
        qbuf, kbuf, vbuf, nidx, abf, BN, N);

    // ---- output projection ----
    gemm_bf16_wmma<<<gblocks, 256, 0, stream>>>(abf, wto, bo, out, BN);
}
